// CrossAttention_43628277792981
// MI455X (gfx1250) — matrix operand order repaired and checked by reading, not compiled, not run on hardware
//
#include <hip/hip_runtime.h>

// ---------------- problem constants ----------------
#define BATCH   2
#define NQ      2048
#define NKV     2048
#define DIM     1024
#define HEADS   16
#define DHEAD   64
#define INNER   1024
#define SCALE_F 0.125f   // 64^-0.5

typedef __attribute__((ext_vector_type(16))) __bf16          v16bf;
typedef __attribute__((ext_vector_type(8)))  float           v8f;
typedef __attribute__((ext_vector_type(8)))  unsigned short  v8u16;
typedef __attribute__((ext_vector_type(4)))  unsigned short  v4u16;
typedef __attribute__((ext_vector_type(4)))  float           v4f;

union Frag { v16bf f; v8u16 u[2]; };

// f32 -> bf16 bits, round-to-nearest-even
static __device__ __forceinline__ unsigned short f2bf(float x) {
  unsigned u = __builtin_bit_cast(unsigned, x);
  u += 0x7fffu + ((u >> 16) & 1u);
  return (unsigned short)(u >> 16);
}

// CDNA5 async global->LDS copy (ASYNCcnt-tracked DMA into LDS).
// VDST carries the LDS byte address (low 32 bits of a generic shared pointer).
static __device__ __forceinline__ void async_b128(unsigned short* lds,
                                                  const unsigned short* g) {
  unsigned l = (unsigned)(unsigned long long)lds;
  asm volatile("global_load_async_to_lds_b128 %0, %1, off"
               :: "v"(l), "v"(g) : "memory");
}
#define WAIT_ASYNC(N) asm volatile("s_wait_asynccnt " #N ::: "memory")
#define WAIT_DS0()    asm volatile("s_wait_dscnt 0x0" ::: "memory")

static __device__ __forceinline__ v8f wmma_bf16(const Frag& a, const Frag& b, v8f c) {
  return __builtin_amdgcn_wmma_f32_16x16x32_bf16(false, a.f, false, b.f,
                                                 (short)0, c, false, false);
}

// 16-lane (half-wave) reductions; wave32 => width 16 splits wave in two groups,
// matching the C/D layout where one matrix row spans lanes {0..15} or {16..31}.
static __device__ __forceinline__ float rmax16(float x) {
  x = fmaxf(x, __shfl_xor(x, 1, 16));
  x = fmaxf(x, __shfl_xor(x, 2, 16));
  x = fmaxf(x, __shfl_xor(x, 4, 16));
  x = fmaxf(x, __shfl_xor(x, 8, 16));
  return x;
}
static __device__ __forceinline__ float rsum16(float x) {
  x += __shfl_xor(x, 1, 16);
  x += __shfl_xor(x, 2, 16);
  x += __shfl_xor(x, 4, 16);
  x += __shfl_xor(x, 8, 16);
  return x;
}

// ---------------------------------------------------------------------------
// Kernel 1: convert + transpose all weights to bf16 [N][K] ("BT") layout.
// ---------------------------------------------------------------------------
__global__ __launch_bounds__(256)
void prep_k(const float* __restrict__ Wq, const float* __restrict__ Wkv,
            const float* __restrict__ Wout,
            unsigned short* __restrict__ WqT, unsigned short* __restrict__ WkT,
            unsigned short* __restrict__ WvT, unsigned short* __restrict__ WoT) {
  const int t = blockIdx.x * 256 + threadIdx.x;     // 1M threads
  const int k = t >> 10, n = t & 1023;
  WqT[(size_t)n * 1024 + k] = f2bf(Wq  [(size_t)k * 1024 + n]);
  WkT[(size_t)n * 1024 + k] = f2bf(Wkv [(size_t)k * 2048 + n]);
  WvT[(size_t)n * 1024 + k] = f2bf(Wkv [(size_t)k * 2048 + 1024 + n]);
  WoT[(size_t)n * 1024 + k] = f2bf(Wout[(size_t)k * 1024 + n]);
}

// ---------------------------------------------------------------------------
// Kernel 2: bf16 WMMA GEMM, C[M=4096][N=1024] = A[M][K=1024] * BT[N][K]^T
//   MODE 0: A=x (f32),   out -> q_bf  [B,H,NQ,64]
//   MODE 1: A=ctx (f32), out -> k_bf  [B,H,NKV,64]
//   MODE 2: A=ctx (f32), out -> vT_bf [B,H,64,NKV]   (transposed store)
//   MODE 3: A=attn (bf16), out -> f32 d_out (+bias)
// Block tile 128x128, BK=64, double-buffered LDS, async B/A staging.
// ---------------------------------------------------------------------------
template<int MODE>
__global__ __launch_bounds__(256, 1)
void gemm_k(const float* __restrict__ Af, const unsigned short* __restrict__ Ab,
            const unsigned short* __restrict__ BT,
            unsigned short* __restrict__ Ob, float* __restrict__ Of,
            const float* __restrict__ bias) {
  constexpr int BM = 128, BN = 128, BK = 64, KD = 1024;
  constexpr int NIT = KD / BK;

  __shared__ __align__(16) unsigned short sA[2][BM * BK];
  __shared__ __align__(16) unsigned short sB[2][BN * BK];

  const int tid  = threadIdx.x;
  const int lane = tid & 31, w = tid >> 5;          // 8 waves (wave32)
  const int r = lane & 15, hl = lane >> 4;
  const int m0 = blockIdx.y * BM, n0 = blockIdx.x * BN;
  const int wm = (w & 3) * 32;                      // wave M offset
  const int wn = (w >> 2) * 64;                     // wave N offset

  const int srow = tid >> 1;                        // staging: 0..127
  const int scol = (tid & 1) * 32;                  // 0 or 32 elements

  v8f acc[2][4];
#pragma unroll
  for (int mi = 0; mi < 2; ++mi)
#pragma unroll
    for (int nj = 0; nj < 4; ++nj) acc[mi][nj] = (v8f)0.0f;

  auto stage = [&](int buf, int kk) {
    {   // B tile: 128x64 bf16, pure async DMA
      const unsigned short* g = BT + (size_t)(n0 + srow) * KD + kk + scol;
      unsigned short* l = &sB[buf][srow * BK + scol];
#pragma unroll
      for (int c = 0; c < 4; ++c) async_b128(l + c * 8, g + c * 8);
    }
    if constexpr (MODE == 3) {   // A already bf16: async DMA
      const unsigned short* g = Ab + (size_t)(m0 + srow) * KD + kk + scol;
      unsigned short* l = &sA[buf][srow * BK + scol];
#pragma unroll
      for (int c = 0; c < 4; ++c) async_b128(l + c * 8, g + c * 8);
    } else {                     // A f32: load, convert, ds_store
      const float* g = Af + (size_t)(m0 + srow) * KD + kk + scol;
      unsigned short* l = &sA[buf][srow * BK + scol];
#pragma unroll
      for (int c = 0; c < 8; ++c) {
        v4f fv = *(const v4f*)(g + c * 4);
        v4u16 h;
        h.x = f2bf(fv.x); h.y = f2bf(fv.y); h.z = f2bf(fv.z); h.w = f2bf(fv.w);
        *(v4u16*)(l + c * 4) = h;
      }
    }
  };

  stage(0, 0);

#pragma unroll 1
  for (int it = 0; it < NIT; ++it) {
    const int buf = it & 1;
    if (it + 1 < NIT) {
      stage(buf ^ 1, (it + 1) * BK);
      // next stage keeps 4 (or 8) async copies in flight; current stage done.
      if constexpr (MODE == 3) { WAIT_ASYNC(0x8); } else { WAIT_ASYNC(0x4); }
    } else {
      WAIT_ASYNC(0x0);
    }
    __syncthreads();

#pragma unroll
    for (int kc = 0; kc < 2; ++kc) {     // two K=32 WMMA steps per BK=64 stage
      Frag a[2], b[4];
#pragma unroll
      for (int mi = 0; mi < 2; ++mi) {   // ISA A layout: K in [8h,8h+8)u[16+8h,..)
        const unsigned short* base = &sA[buf][(wm + mi * 16 + r) * BK + kc * 32];
        a[mi].u[0] = *(const v8u16*)(base + 8 * hl);
        a[mi].u[1] = *(const v8u16*)(base + 16 + 8 * hl);
      }
#pragma unroll
      for (int nj = 0; nj < 4; ++nj) {   // ISA B layout: K in [8h,8h+8)u[16+8h,..)
        const unsigned short* base =
            &sB[buf][(wn + nj * 16 + r) * BK + kc * 32 + 8 * hl];
        b[nj].u[0] = *(const v8u16*)(base);
        b[nj].u[1] = *(const v8u16*)(base + 16);
      }
#pragma unroll
      for (int mi = 0; mi < 2; ++mi)
#pragma unroll
        for (int nj = 0; nj < 4; ++nj)
          acc[mi][nj] = wmma_bf16(a[mi], b[nj], acc[mi][nj]);
    }
    __syncthreads();
  }

  // Epilogue: C layout => element (VGPR i, lane) is (M = i + 8*hl, N = r)
#pragma unroll
  for (int mi = 0; mi < 2; ++mi) {
#pragma unroll
    for (int nj = 0; nj < 4; ++nj) {
#pragma unroll
      for (int i = 0; i < 8; ++i) {
        const int m = m0 + wm + mi * 16 + i + 8 * hl;
        const int n = n0 + wn + nj * 16 + r;
        const float v = acc[mi][nj][i];
        if constexpr (MODE == 3) {
          Of[(size_t)m * DIM + n] = v + bias[n];
        } else {
          const int bb = m >> 11, t = m & 2047;  // (batch, token)
          const int h = n >> 6,  d = n & 63;     // (head, dim)
          if constexpr (MODE == 2) {             // V stored transposed
            Ob[((size_t)(bb * HEADS + h) * DHEAD + d) * NKV + t] = f2bf(v);
          } else {
            Ob[((size_t)(bb * HEADS + h) * NQ + t) * DHEAD + d] = f2bf(v);
          }
        }
      }
    }
  }
}

// ---------------------------------------------------------------------------
// Kernel 3: flash attention. Block = 4 waves, 64 q-rows; kv tiles of 64.
// ---------------------------------------------------------------------------
__global__ __launch_bounds__(128, 1)
void attn_k(const unsigned short* __restrict__ Qb,
            const unsigned short* __restrict__ Kb,
            const unsigned short* __restrict__ Vt,
            unsigned short* __restrict__ Ob) {
  __shared__ __align__(16) unsigned short sK[64 * 64];
  __shared__ __align__(16) unsigned short sV[64 * 64];
  __shared__ __align__(16) unsigned short sP[4][16 * 64];

  const int tid  = threadIdx.x;
  const int lane = tid & 31, w = tid >> 5;
  const int r = lane & 15, hl = lane >> 4;
  const int bh = blockIdx.y;
  const int q0 = blockIdx.x * 64;

  const unsigned short* Qh = Qb + (size_t)bh * NQ * DHEAD;
  const unsigned short* Kh = Kb + (size_t)bh * NKV * DHEAD;
  const unsigned short* Vh = Vt + (size_t)bh * DHEAD * NKV;

  // Q fragments (kept in registers for the whole kv sweep)
  Frag qa[2];
  {
    const unsigned short* qrow = Qh + (size_t)(q0 + w * 16 + r) * DHEAD;
#pragma unroll
    for (int kc = 0; kc < 2; ++kc) {
      qa[kc].u[0] = *(const v8u16*)(qrow + kc * 32 + 8 * hl);
      qa[kc].u[1] = *(const v8u16*)(qrow + kc * 32 + 16 + 8 * hl);
    }
  }

  float mrow[8], lrow[8];
  v8f o[4];
#pragma unroll
  for (int i = 0; i < 8; ++i) { mrow[i] = -3.0e38f; lrow[i] = 0.0f; }
#pragma unroll
  for (int dj = 0; dj < 4; ++dj) o[dj] = (v8f)0.0f;

  const int srow = tid >> 1;          // 0..63
  const int scol = (tid & 1) * 32;    // 0 / 32 elements

#pragma unroll 1
  for (int kv0 = 0; kv0 < NKV; kv0 += 64) {
    {  // async-stage K tile [kv][d] and Vt tile [d][kv]
      const unsigned short* gk = Kh + (size_t)(kv0 + srow) * DHEAD + scol;
      unsigned short* lk = &sK[srow * 64 + scol];
#pragma unroll
      for (int c = 0; c < 4; ++c) async_b128(lk + c * 8, gk + c * 8);
      const unsigned short* gv = Vh + (size_t)srow * NKV + kv0 + scol;
      unsigned short* lv = &sV[srow * 64 + scol];
#pragma unroll
      for (int c = 0; c < 4; ++c) async_b128(lv + c * 8, gv + c * 8);
    }
    WAIT_ASYNC(0x0);
    __syncthreads();

    // S = Q * K^T  (K stored [kv][d] is exactly the BT layout, N=kv, K=d)
    v8f s[4];
#pragma unroll
    for (int nj = 0; nj < 4; ++nj) {
      Frag b0, b1;
      const unsigned short* base = &sK[(nj * 16 + r) * 64];
      b0.u[0] = *(const v8u16*)(base + 8 * hl);
      b0.u[1] = *(const v8u16*)(base + 16 + 8 * hl);
      b1.u[0] = *(const v8u16*)(base + 32 + 8 * hl);
      b1.u[1] = *(const v8u16*)(base + 32 + 16 + 8 * hl);
      v8f z = (v8f)0.0f;
      z = wmma_bf16(qa[0], b0, z);
      z = wmma_bf16(qa[1], b1, z);
      s[nj] = z;
    }

    // online softmax (rows live across 16-lane half-waves)
    float mnew[8], rs[8];
#pragma unroll
    for (int i = 0; i < 8; ++i) {
      float xm = fmaxf(fmaxf(s[0][i], s[1][i]), fmaxf(s[2][i], s[3][i])) * SCALE_F;
      xm = rmax16(xm);
      mnew[i] = fmaxf(mrow[i], xm);
      rs[i] = 0.0f;
    }
#pragma unroll
    for (int nj = 0; nj < 4; ++nj) {
#pragma unroll
      for (int i = 0; i < 8; ++i) {
        const float p = __expf(s[nj][i] * SCALE_F - mnew[i]);
        rs[i] += p;
        sP[w][(i + 8 * hl) * 64 + nj * 16 + r] = f2bf(p);  // C -> A relayout via LDS
      }
    }
#pragma unroll
    for (int i = 0; i < 8; ++i) {
      const float rsum = rsum16(rs[i]);
      const float sc = __expf(mrow[i] - mnew[i]);
      lrow[i] = lrow[i] * sc + rsum;
      mrow[i] = mnew[i];
#pragma unroll
      for (int dj = 0; dj < 4; ++dj) o[dj][i] *= sc;
    }
    WAIT_DS0();  // P writes are wave-local: DScnt wait suffices, no barrier

    // O += P * V   (Vt[d][kv] is BT layout, N=d, K=kv)
    Frag pa[2];
    {
      const unsigned short* prow = &sP[w][r * 64];
#pragma unroll
      for (int kc = 0; kc < 2; ++kc) {
        pa[kc].u[0] = *(const v8u16*)(prow + kc * 32 + 8 * hl);
        pa[kc].u[1] = *(const v8u16*)(prow + kc * 32 + 16 + 8 * hl);
      }
    }
#pragma unroll
    for (int dj = 0; dj < 4; ++dj) {
      Frag b0, b1;
      const unsigned short* base = &sV[(dj * 16 + r) * 64];
      b0.u[0] = *(const v8u16*)(base + 8 * hl);
      b0.u[1] = *(const v8u16*)(base + 16 + 8 * hl);
      b1.u[0] = *(const v8u16*)(base + 32 + 8 * hl);
      b1.u[1] = *(const v8u16*)(base + 32 + 16 + 8 * hl);
      o[dj] = wmma_bf16(pa[0], b0, o[dj]);
      o[dj] = wmma_bf16(pa[1], b1, o[dj]);
    }
    __syncthreads();  // protect sK/sV before next stage overwrites
  }

  const int bb = bh / HEADS, h = bh % HEADS;
#pragma unroll
  for (int dj = 0; dj < 4; ++dj) {
#pragma unroll
    for (int i = 0; i < 8; ++i) {
      const int row = q0 + w * 16 + i + 8 * hl;
      const int col = h * DHEAD + dj * 16 + r;
      const float v = o[dj][i] / lrow[i];
      Ob[(size_t)(bb * NQ + row) * INNER + col] = f2bf(v);
    }
  }
}

// ---------------------------------------------------------------------------
extern "C" void kernel_launch(void* const* d_in, const int* in_sizes, int n_in,
                              void* d_out, int out_size, void* d_ws, size_t ws_size,
                              hipStream_t stream) {
  const float* x    = (const float*)d_in[0];
  const float* ctx  = (const float*)d_in[1];
  const float* Wq   = (const float*)d_in[2];
  const float* Wkv  = (const float*)d_in[3];
  const float* Wout = (const float*)d_in[4];
  const float* bout = (const float*)d_in[5];
  float* out = (float*)d_out;

  char* ws = (char*)d_ws;
  size_t off = 0;
  auto take = [&](size_t bytes) {
    void* p = ws + off;
    off += (bytes + 255) & ~(size_t)255;
    return p;
  };
  unsigned short* WqT = (unsigned short*)take((size_t)DIM * INNER * 2);
  unsigned short* WkT = (unsigned short*)take((size_t)DIM * INNER * 2);
  unsigned short* WvT = (unsigned short*)take((size_t)DIM * INNER * 2);
  unsigned short* WoT = (unsigned short*)take((size_t)DIM * INNER * 2);
  unsigned short* qb  = (unsigned short*)take((size_t)BATCH * NQ  * INNER * 2);
  unsigned short* kb  = (unsigned short*)take((size_t)BATCH * NKV * INNER * 2);
  unsigned short* vtb = (unsigned short*)take((size_t)BATCH * NKV * INNER * 2);
  unsigned short* ab  = (unsigned short*)take((size_t)BATCH * NQ  * INNER * 2);

  prep_k<<<4096, 256, 0, stream>>>(Wq, Wkv, Wout, WqT, WkT, WvT, WoT);

  const dim3 ggrid(INNER / 128, (BATCH * NQ) / 128);   // (8, 32)
  gemm_k<0><<<ggrid, 256, 0, stream>>>(x,   nullptr, WqT, qb,  nullptr, nullptr);
  gemm_k<1><<<ggrid, 256, 0, stream>>>(ctx, nullptr, WkT, kb,  nullptr, nullptr);
  gemm_k<2><<<ggrid, 256, 0, stream>>>(ctx, nullptr, WvT, vtb, nullptr, nullptr);

  attn_k<<<dim3(NQ / 64, BATCH * HEADS), 128, 0, stream>>>(qb, kb, vtb, ab);

  gemm_k<3><<<ggrid, 256, 0, stream>>>(nullptr, ab, WoT, nullptr, out, bout);
}
